// EnhancedIsotropyMaximizationLoss_78108275245360
// MI455X (gfx1250) — matrix operand order repaired and checked by reading, not compiled, not run on hardware
//
#include <hip/hip_runtime.h>
#include <math.h>

#define DCOLS 256
#define EPSF 1e-6f

typedef __bf16 bf16_t;
typedef bf16_t v16bf __attribute__((ext_vector_type(16)));
typedef float  v8f   __attribute__((ext_vector_type(8)));
typedef unsigned int v4u __attribute__((ext_vector_type(4)));

union Frag { v16bf v; v4u q[2]; };

static __device__ __forceinline__ unsigned short f32_to_bf16_rne(float f) {
    unsigned int u = __float_as_uint(f);
    u += 0x7FFFu + ((u >> 16) & 1u);   // round to nearest even
    return (unsigned short)(u >> 16);
}
static __device__ __forceinline__ float bf16_to_f32(unsigned short h) {
    return __uint_as_float(((unsigned int)h) << 16);
}

// CDNA5 async global->LDS copy (ASYNCcnt-tracked), 16 bytes per lane.
// ldsoff = LDS byte offset (low 32 bits of generic shared pointer, per ISA aperture rules).
static __device__ __forceinline__ void async_copy_b128(unsigned ldsoff, unsigned long long gaddr) {
    asm volatile("global_load_async_to_lds_b128 %0, %1, off"
                 :: "v"(ldsoff), "v"(gaddr) : "memory");
}
static __device__ __forceinline__ void wait_asynccnt0() {
    asm volatile("s_wait_asynccnt 0x0" ::: "memory");
}

// ---------------- Kernel 1: L2-normalize rows ----------------
__global__ void k_normalize(const float* __restrict__ x, float* __restrict__ e) {
    int row = blockIdx.x;
    int t   = threadIdx.x;               // 256 threads = D columns
    float v = x[(size_t)row * DCOLS + t];
    __shared__ float red[256];
    red[t] = v * v;
    __syncthreads();
    for (int s = 128; s > 0; s >>= 1) { if (t < s) red[t] += red[t + s]; __syncthreads(); }
    float inv = 1.0f / (sqrtf(red[0]) + EPSF);
    e[(size_t)row * DCOLS + t] = v * inv;
}

// ---------------- Kernel 2: column means ----------------
__global__ void k_colmean(const float* __restrict__ e, float* __restrict__ colmean, int Nrows) {
    int col = blockIdx.x;                // one block per column
    int t   = threadIdx.x;               // 256 threads
    float s = 0.f;
    for (int r = t; r < Nrows; r += 256) s += e[(size_t)r * DCOLS + col];
    __shared__ float red[256];
    red[t] = s;
    __syncthreads();
    for (int k = 128; k > 0; k >>= 1) { if (t < k) red[t] += red[t + k]; __syncthreads(); }
    if (t == 0) colmean[col] = red[0] / (float)Nrows;
}

// ------- Kernel 3: center, bf16x2 split, per-row squared norms -------
__global__ void k_center_split(const float* __restrict__ e, const float* __restrict__ colmean,
                               unsigned short* __restrict__ chi, unsigned short* __restrict__ clo,
                               float* __restrict__ sqn) {
    int row = blockIdx.x;
    int t   = threadIdx.x;
    float c  = e[(size_t)row * DCOLS + t] - colmean[t];
    unsigned short hb = f32_to_bf16_rne(c);
    float hf = bf16_to_f32(hb);
    unsigned short lb = f32_to_bf16_rne(c - hf);
    chi[(size_t)row * DCOLS + t] = hb;
    clo[(size_t)row * DCOLS + t] = lb;
    __shared__ float red[256];
    red[t] = c * c;
    __syncthreads();
    for (int s = 128; s > 0; s >>= 1) { if (t < s) red[t] += red[t + s]; __syncthreads(); }
    if (t == 0) sqn[row] = red[0];
}

// ------- Kernel 4: fused split-bf16 WMMA Gram + distance reduction -------
// One block per upper-triangular 128x128 tile pair (i<=j); 256 threads = 8 waves.
// B tile k-chunks staged once per block into LDS via async copies (double buffered);
// each wave computes a 16x128 strip with 3 WMMAs (hi*hi + hi*lo + lo*hi) per n-tile.
__global__ void k_gram_dist(const unsigned short* __restrict__ chi,
                            const unsigned short* __restrict__ clo,
                            const float* __restrict__ sqn,
                            float* __restrict__ partial, int T) {
    // LDS B staging: [buf:2][mat:2][nt:8][lane:32][16 shorts(32B frag)] = 32 KB
    __shared__ __align__(16) unsigned short ldsB[2 * 2 * 8 * 32 * 16];
    __shared__ float red[256];

    int p = blockIdx.x;
    int i = 0, rem = p;
    while (rem >= T - i) { rem -= (T - i); ++i; }
    int j = i + rem;

    int wave = threadIdx.x >> 5;
    int lane = threadIdx.x & 31;
    int half = lane >> 4;          // 0: lanes 0-15, 1: lanes 16-31
    int l15  = lane & 15;

    int rowBase = i * 128 + wave * 16;
    int colBase = j * 128;

    const unsigned short* Ah = chi + (size_t)(rowBase + l15) * DCOLS;
    const unsigned short* Al = clo + (size_t)(rowBase + l15) * DCOLS;
    const int aoff = half * 16;    // A frag: 32 contiguous bytes at k0+aoff
    const int boff = half * 16;    // B frag: 32 contiguous bytes at k0+boff

    // Staging role: thread (nt = tid>>5, ln = tid&31) fills the frag that
    // consumer lane `ln` of n-tile `nt` will read.  32 B per matrix.
    const unsigned ldsBase = (unsigned)(size_t)&ldsB[0];
    const int snt = wave;                 // tid>>5
    const size_t srow = (size_t)(colBase + snt * 16 + l15) * DCOLS + boff;
    const unsigned sdst = ldsBase + (unsigned)((snt * 32 + lane) * 32);

    v8f acc[8];
#pragma unroll
    for (int nt = 0; nt < 8; ++nt) acc[nt] = (v8f){0.f,0.f,0.f,0.f,0.f,0.f,0.f,0.f};

    // Prologue: stage k-chunk 0 into buffer 0
    {
        unsigned long long gh = (unsigned long long)(size_t)(chi + srow);
        unsigned long long gl = (unsigned long long)(size_t)(clo + srow);
        async_copy_b128(sdst,              gh);
        async_copy_b128(sdst + 16,         gh + 16);
        async_copy_b128(sdst + 8192,       gl);          // mat stride = 8 KB
        async_copy_b128(sdst + 8192 + 16,  gl + 16);
    }
    wait_asynccnt0();
    __syncthreads();

    for (int ks = 0; ks < DCOLS / 32; ++ks) {
        const int k0  = ks * 32;
        const int cur = ks & 1;
        // Prefetch next k-chunk into the other buffer (overlaps with WMMAs below)
        if (ks + 1 < DCOLS / 32) {
            unsigned nb = (unsigned)(((ks + 1) & 1) * 16384);   // buf stride = 16 KB
            unsigned long long gh = (unsigned long long)(size_t)(chi + srow + (k0 + 32));
            unsigned long long gl = (unsigned long long)(size_t)(clo + srow + (k0 + 32));
            async_copy_b128(sdst + nb,             gh);
            async_copy_b128(sdst + nb + 16,        gh + 16);
            async_copy_b128(sdst + nb + 8192,      gl);
            async_copy_b128(sdst + nb + 8192 + 16, gl + 16);
        }
        // A fragments: direct global loads (wave-private rows, no duplication)
        Frag ah, al;
        ah.q[0] = *(const v4u*)(Ah + k0 + aoff);
        ah.q[1] = *(const v4u*)(Ah + k0 + aoff + 8);
        al.q[0] = *(const v4u*)(Al + k0 + aoff);
        al.q[1] = *(const v4u*)(Al + k0 + aoff + 8);
#pragma unroll
        for (int nt = 0; nt < 8; ++nt) {
            const unsigned short* bp = ldsB + cur * 8192 + (nt * 32 + lane) * 16; // shorts
            Frag bh, bl;
            bh.q[0] = *(const v4u*)(bp);
            bh.q[1] = *(const v4u*)(bp + 8);
            bl.q[0] = *(const v4u*)(bp + 4096);          // lo matrix, +8 KB
            bl.q[1] = *(const v4u*)(bp + 4096 + 8);
            // gram ~= hi*lo + lo*hi + hi*hi  (fp32 accumulate)
            acc[nt] = __builtin_amdgcn_wmma_f32_16x16x32_bf16(false, ah.v, false, bl.v,
                                                              (short)0, acc[nt], false, false);
            acc[nt] = __builtin_amdgcn_wmma_f32_16x16x32_bf16(false, al.v, false, bh.v,
                                                              (short)0, acc[nt], false, false);
            acc[nt] = __builtin_amdgcn_wmma_f32_16x16x32_bf16(false, ah.v, false, bh.v,
                                                              (short)0, acc[nt], false, false);
        }
        wait_asynccnt0();       // our prefetch for next buffer has landed in LDS
        __syncthreads();        // everyone done reading `cur`; publish next buffer
    }

    // Epilogue: sq_dist = ||a||^2 + ||b||^2 - 2*gram; sum sqrt.
    float sqr[8];
#pragma unroll
    for (int v = 0; v < 8; ++v) sqr[v] = sqn[rowBase + v + half * 8];

    float lsum = 0.f;
#pragma unroll
    for (int nt = 0; nt < 8; ++nt) {
        int cg = colBase + nt * 16 + l15;
        float sc = sqn[cg];
#pragma unroll
        for (int v = 0; v < 8; ++v) {
            int rg = rowBase + v + half * 8;
            float d2 = sqr[v] + sc - 2.0f * acc[nt][v];
            float d  = (d2 > 0.f) ? sqrtf(d2) : 0.f;
            if (rg == cg) d = 0.f;   // diagonal distance is exactly 0
            lsum += d;
        }
    }

    red[threadIdx.x] = lsum;
    __syncthreads();
    for (int s = 128; s > 0; s >>= 1) {
        if (threadIdx.x < s) red[threadIdx.x] += red[threadIdx.x + s];
        __syncthreads();
    }
    if (threadIdx.x == 0) partial[p] = (i == j) ? red[0] : 2.0f * red[0];
}

// ---------------- Kernel 5: final deterministic reduction ----------------
__global__ void k_final(const float* __restrict__ partial, int nPart,
                        const float* __restrict__ sqn, int Nrows, float* __restrict__ out) {
    int t = threadIdx.x;
    __shared__ float red[256];

    float s = 0.f;
    for (int q = t; q < nPart; q += 256) s += partial[q];
    red[t] = s;
    __syncthreads();
    for (int k = 128; k > 0; k >>= 1) { if (t < k) red[t] += red[t + k]; __syncthreads(); }
    float distSum = red[0];
    __syncthreads();

    float s2 = 0.f;
    for (int r = t; r < Nrows; r += 256) s2 += sqn[r];
    red[t] = s2;
    __syncthreads();
    for (int k = 128; k > 0; k >>= 1) { if (t < k) red[t] += red[t + k]; __syncthreads(); }

    if (t == 0) {
        float totalcc = red[0];                       // sum(c*c)
        float diag_var_mean = totalcc / (float)DCOLS;
        double meanD = (double)distSum / ((double)Nrows * (double)Nrows);
        out[0] = 1.0f / diag_var_mean + logf((float)meanD);
    }
}

extern "C" void kernel_launch(void* const* d_in, const int* in_sizes, int n_in,
                              void* d_out, int out_size, void* d_ws, size_t ws_size,
                              hipStream_t stream) {
    (void)n_in; (void)out_size; (void)ws_size;
    const float* x = (const float*)d_in[0];
    int Nrows = in_sizes[0] / DCOLS;                  // 8192
    float* out = (float*)d_out;

    char* ws = (char*)d_ws;
    size_t off = 0;
    float* e = (float*)(ws + off);               off += (size_t)Nrows * DCOLS * sizeof(float);
    unsigned short* chi = (unsigned short*)(ws + off); off += (size_t)Nrows * DCOLS * 2;
    unsigned short* clo = (unsigned short*)(ws + off); off += (size_t)Nrows * DCOLS * 2;
    float* colmean = (float*)(ws + off);         off += DCOLS * sizeof(float);
    float* sqn = (float*)(ws + off);             off += (size_t)Nrows * sizeof(float);
    float* partial = (float*)(ws + off);

    int T = Nrows / 128;                              // 64 tiles
    int nPart = T * (T + 1) / 2;                      // 2080 tile pairs

    k_normalize   <<<Nrows, 256, 0, stream>>>(x, e);
    k_colmean     <<<DCOLS, 256, 0, stream>>>(e, colmean, Nrows);
    k_center_split<<<Nrows, 256, 0, stream>>>(e, colmean, chi, clo, sqn);
    k_gram_dist   <<<nPart, 256, 0, stream>>>(chi, clo, sqn, partial, T);
    k_final       <<<1,     256, 0, stream>>>(partial, nPart, sqn, Nrows, out);
}
